// GrammarTreeDecoder_63015760167367
// MI455X (gfx1250) — matrix operand order repaired and checked by reading, not compiled, not run on hardware
//
#include <hip/hip_runtime.h>
#include <cmath>

#define NN 4096
#define HH 256
#define EE 128
#define CC 32
#define VV 64
#define KK 20
#define GG (4*HH)       // 1024 gate outputs
#define OSTR (1+2*HH)   // 513 output stride
#define MT (NN/16)      // worst-case 16-row tiles per bucket (logits)
#define MT32 (NN/32)    // worst-case 32-row tiles per bucket (lstm)

typedef __attribute__((ext_vector_type(16))) __bf16 v16bf;
typedef __attribute__((ext_vector_type(8)))  float  v8f;
typedef __attribute__((ext_vector_type(4)))  int    v4i;

// workspace layout (int units)
#define CNTK_OFF 0
#define CNTC_OFF 32
#define IDXK_OFF 64
#define IDXC_OFF (64 + KK*NN)

// ---- async global->LDS (gfx1250) with safe fallback ----
#if defined(__has_builtin)
#  if __has_builtin(__builtin_amdgcn_global_load_async_to_lds_b128)
#    define HAVE_ASYNC_LDS 1
#  endif
#endif

typedef __attribute__((address_space(1))) v4i as1_v4i;
typedef __attribute__((address_space(3))) v4i as3_v4i;

__device__ __forceinline__ void async_copy16(const float* gsrc, float* ldst) {
#ifdef HAVE_ASYNC_LDS
    __builtin_amdgcn_global_load_async_to_lds_b128(
        (as1_v4i*)gsrc, (as3_v4i*)ldst, 0, 0);
#else
    *(float4*)ldst = *(const float4*)gsrc;
#endif
}

__device__ __forceinline__ void async_wait_all() {
#ifdef HAVE_ASYNC_LDS
#  if __has_builtin(__builtin_amdgcn_s_wait_asynccnt)
    __builtin_amdgcn_s_wait_asynccnt(0);
#  else
    asm volatile("s_wait_asynccnt 0" ::: "memory");
#  endif
#endif
}

// ---- WMMA fragment lane layouts (CDNA5 ISA 7.12.2) ----
// A 16-bit 16x32: lanes 0-15 M=lane, K={0..7,16..23}; lanes 16-31 M=lane-16, K={8..15,24..31}
__device__ __forceinline__ int klocA(int j, int lane) {
    int v = j >> 1, p = j & 1;
    int base = (lane < 16) ? 0 : 8;
    return (v < 4) ? (2*v + p + base) : (16 + 2*(v-4) + p + base);
}
// B 16-bit 32x16: lanes 0-15 N=lane, K={0..7,16..23}; lanes 16-31 N=lane-16, K={8..15,24..31}
__device__ __forceinline__ int klocB(int j, int lane) {
    int base = (lane < 16) ? 0 : 8;
    return (j < 8) ? (j + base) : (16 + (j - 8) + base);
}

__device__ __forceinline__ v8f wmma_bf16(v16bf a, v16bf b, v8f c) {
    return __builtin_amdgcn_wmma_f32_16x16x32_bf16(
        false, a, false, b, (short)0, c, false, false);
}

__device__ __forceinline__ float sigf(float x) { return 1.f / (1.f + __expf(-x)); }

// ---------------- bucket build ----------------
__global__ void k_zero(int* ws) {
    if (threadIdx.x < 64) ws[threadIdx.x] = 0;
}

__global__ void k_bucket(const int* __restrict__ cell_id,
                         const int* __restrict__ category, int* ws) {
    int n = blockIdx.x * 256 + threadIdx.x;
    if (n >= NN) return;
    int k = cell_id[n];
    int r = atomicAdd(&ws[CNTK_OFF + k], 1);
    ws[IDXK_OFF + k*NN + r] = n;
    int c = category[n];
    int r2 = atomicAdd(&ws[CNTC_OFF + c], 1);
    ws[IDXC_OFF + c*NN + r2] = n;
}

// ---------------- category head: logits + CE ----------------
__global__ void k_logits(const float* __restrict__ vec,
                         const float* __restrict__ Wlin,
                         const float* __restrict__ blin,
                         const int*   __restrict__ true_idx,
                         const int*   __restrict__ ws,
                         float* __restrict__ out) {
    int c = blockIdx.x / MT;
    int t = blockIdx.x % MT;
    int cnt = ws[CNTC_OFF + c];
    if (t*16 >= cnt) return;

    int lane = threadIdx.x;
    int pos  = t*16 + (lane & 15);
    int posc = min(pos, cnt - 1);
    int row  = ws[IDXC_OFF + c*NN + posc];

    v8f acc[4];
#pragma unroll
    for (int ct = 0; ct < 4; ct++) acc[ct] = (v8f){0,0,0,0,0,0,0,0};

    const float* vrow = vec  + row * HH;
    const float* Wc   = Wlin + c * VV * HH;
    for (int cc = 0; cc < 8; cc++) {           // H=256 / 32
        v16bf a;
#pragma unroll
        for (int j = 0; j < 16; j++) a[j] = (__bf16)vrow[cc*32 + klocA(j, lane)];
#pragma unroll
        for (int ct = 0; ct < 4; ct++) {
            int col = ct*16 + (lane & 15);
            const float* wb = Wc + col * HH + cc*32;
            v16bf b;
#pragma unroll
            for (int j = 0; j < 16; j++) b[j] = (__bf16)wb[klocB(j, lane)];
            acc[ct] = wmma_bf16(a, b, acc[ct]);
        }
    }

    __shared__ float lg[16][VV];
    int nloc  = lane & 15;
    int mbase = (lane < 16) ? 0 : 8;
#pragma unroll
    for (int ct = 0; ct < 4; ct++) {
        float bl = blin[c*VV + ct*16 + nloc];
#pragma unroll
        for (int r = 0; r < 8; r++)
            lg[mbase + r][ct*16 + nloc] = acc[ct][r] + bl;
    }
    __syncthreads();

    if (lane < 16 && (t*16 + lane) < cnt) {
        int rown = ws[IDXC_OFF + c*NN + t*16 + lane];
        float mx = -1e30f;
        for (int v = 0; v < VV; v++) mx = fmaxf(mx, lg[lane][v]);
        float s = 0.f;
        for (int v = 0; v < VV; v++) s += __expf(lg[lane][v] - mx);
        float lse = mx + __logf(s);
        out[rown * OSTR] = lse - lg[lane][true_idx[rown]];
    }
}

// ---------------- tiny ec/rel heads: CE accumulate ----------------
__global__ void k_ecrel(const float* __restrict__ vec,
                        const float* __restrict__ Wec,  const float* __restrict__ bec,
                        const float* __restrict__ Wrel, const float* __restrict__ brel,
                        const int* __restrict__ ec_idx, const int* __restrict__ rel_idx,
                        float* __restrict__ out) {
    int w = threadIdx.x >> 5;
    int lane = threadIdx.x & 31;
    int n = blockIdx.x * 8 + w;
    if (n >= NN) return;
    float acc[7] = {0,0,0,0,0,0,0};
    const float* vrow = vec + n * HH;
    for (int h = lane; h < HH; h += 32) {
        float v = vrow[h];
        acc[0] += v * Wec[h];
        acc[1] += v * Wec[HH + h];
#pragma unroll
        for (int j = 0; j < 5; j++) acc[2 + j] += v * Wrel[j*HH + h];
    }
#pragma unroll
    for (int j = 0; j < 7; j++)
        for (int off = 16; off; off >>= 1) acc[j] += __shfl_xor(acc[j], off, 32);
    if (lane == 0) {
        float a0 = acc[0] + bec[0], a1 = acc[1] + bec[1];
        float m  = fmaxf(a0, a1);
        float lse = m + __logf(__expf(a0 - m) + __expf(a1 - m));
        float le  = lse - (ec_idx[n] == 0 ? a0 : a1);
        float rv[5]; float rm = -1e30f;
#pragma unroll
        for (int j = 0; j < 5; j++) { rv[j] = acc[2 + j] + brel[j]; rm = fmaxf(rm, rv[j]); }
        float rs = 0.f;
#pragma unroll
        for (int j = 0; j < 5; j++) rs += __expf(rv[j] - rm);
        float lr = (rm + __logf(rs)) - rv[rel_idx[n]];
        out[n * OSTR] += le + lr;
    }
}

// ---------------- bucketed LSTM cell: 32-row WMMA tiles + fused pointwise ----------------
__global__ void __launch_bounds__(256)
k_lstm(const float* __restrict__ inp,  const float* __restrict__ hprev,
       const float* __restrict__ cprev,
       const float* __restrict__ W_ih, const float* __restrict__ b_ih,
       const float* __restrict__ W_hh, const float* __restrict__ b_hh,
       const int*   __restrict__ ws,   float* __restrict__ out) {
    int k = blockIdx.x / MT32;
    int t = blockIdx.x % MT32;
    int cnt = ws[CNTK_OFF + k];
    if (t*32 >= cnt) return;

    __shared__ int rows_s[32];
    // 64 KB region, time-shared: phase 1 = staged X (f32 [32][384], 48 KB),
    // phase 2 = gates (bf16 [32][1024], 64 KB)
    __shared__ __align__(16) char smem[32 * GG * 2];
    float  (*Xs)[384] = (float (*)[384])smem;
    __bf16 (*gs)[GG]  = (__bf16 (*)[GG])smem;

    int tid = threadIdx.x;
    if (tid < 32) {
        int pos = min(t*32 + tid, cnt - 1);
        rows_s[tid] = ws[IDXK_OFF + k*NN + pos];
    }
    __syncthreads();

    // stage gathered [inp | h_prev] rows into LDS (async b128 per lane: 16B each)
    {
        int w = tid >> 5, lane = tid & 31;
#pragma unroll
        for (int i = 0; i < 4; i++) {
            int m = w*4 + i;
            int row = rows_s[m];
            async_copy16(inp   + row*EE + lane*4,        &Xs[m][lane*4]);
            async_copy16(hprev + row*HH + lane*4,        &Xs[m][EE + lane*4]);
            async_copy16(hprev + row*HH + EE + lane*4,   &Xs[m][2*EE + lane*4]);
        }
        async_wait_all();
    }
    __syncthreads();

    int w = tid >> 5, lane = tid & 31;
    int m = lane & 15, nloc = lane & 15;
    v8f acc0[8], acc1[8];
#pragma unroll
    for (int ct = 0; ct < 8; ct++) {
        acc0[ct] = (v8f){0,0,0,0,0,0,0,0};
        acc1[ct] = (v8f){0,0,0,0,0,0,0,0};
    }

    const float* Wih_k = W_ih + (size_t)k * GG * EE;
    const float* Whh_k = W_hh + (size_t)k * GG * HH;
    for (int cc = 0; cc < 12; cc++) {          // 384 / 32 k-steps
        v16bf a0, a1;
#pragma unroll
        for (int j = 0; j < 16; j++) {
            int kk = cc*32 + klocA(j, lane);
            a0[j] = (__bf16)Xs[m][kk];
            a1[j] = (__bf16)Xs[16 + m][kk];
        }
#pragma unroll
        for (int ct = 0; ct < 8; ct++) {
            int col = w*128 + ct*16 + nloc;
            v16bf b;
            if (cc < 4) {
                const float* wb = Wih_k + col*EE + cc*32;
#pragma unroll
                for (int j = 0; j < 16; j++) b[j] = (__bf16)wb[klocB(j, lane)];
            } else {
                const float* wb = Whh_k + col*HH + (cc - 4)*32;
#pragma unroll
                for (int j = 0; j < 16; j++) b[j] = (__bf16)wb[klocB(j, lane)];
            }
            acc0[ct] = wmma_bf16(a0, b, acc0[ct]);   // rows 0-15 share this B
            acc1[ct] = wmma_bf16(a1, b, acc1[ct]);   // rows 16-31 reuse it
        }
    }

    __syncthreads();   // everyone done reading Xs; smem region becomes gates

    int mbase = (lane < 16) ? 0 : 8;
#pragma unroll
    for (int ct = 0; ct < 8; ct++) {
        int col = w*128 + ct*16 + nloc;
        float bsum = b_ih[k*GG + col] + b_hh[k*GG + col];
#pragma unroll
        for (int r = 0; r < 8; r++) {
            gs[mbase + r][col]      = (__bf16)(acc0[ct][r] + bsum);
            gs[16 + mbase + r][col] = (__bf16)(acc1[ct][r] + bsum);
        }
    }
    __syncthreads();

    // fused pointwise LSTM + scatter to output
    for (int e = tid; e < 32*HH; e += 256) {
        int mm = e >> 8;
        int h  = e & 255;
        if (t*32 + mm < cnt) {
            int row = rows_s[mm];
            float gi = (float)gs[mm][h];
            float gf = (float)gs[mm][HH + h];
            float gg = (float)gs[mm][2*HH + h];
            float go = (float)gs[mm][3*HH + h];
            float cp = cprev[row*HH + h];
            float cn = sigf(gf)*cp + sigf(gi)*tanhf(gg);
            float hn = sigf(go)*tanhf(cn);
            out[row*OSTR + 1 + h]      = hn;
            out[row*OSTR + 1 + HH + h] = cn;
        }
    }
}

extern "C" void kernel_launch(void* const* d_in, const int* in_sizes, int n_in,
                              void* d_out, int out_size, void* d_ws, size_t ws_size,
                              hipStream_t stream) {
    const float* vec    = (const float*)d_in[0];
    const float* inp    = (const float*)d_in[1];
    const float* h_prev = (const float*)d_in[2];
    const float* c_prev = (const float*)d_in[3];
    const float* Wlin   = (const float*)d_in[4];
    const float* blin   = (const float*)d_in[5];
    const float* Wec    = (const float*)d_in[6];
    const float* bec    = (const float*)d_in[7];
    const float* Wrel   = (const float*)d_in[8];
    const float* brel   = (const float*)d_in[9];
    const float* W_ih   = (const float*)d_in[10];
    const float* b_ih   = (const float*)d_in[11];
    const float* W_hh   = (const float*)d_in[12];
    const float* b_hh   = (const float*)d_in[13];
    const int* category = (const int*)d_in[14];
    const int* cell_id  = (const int*)d_in[15];
    const int* true_idx = (const int*)d_in[16];
    const int* ec_idx   = (const int*)d_in[17];
    const int* rel_idx  = (const int*)d_in[18];
    float* out = (float*)d_out;
    int*   wsI = (int*)d_ws;

    k_zero  <<<1, 64, 0, stream>>>(wsI);
    k_bucket<<<NN/256, 256, 0, stream>>>(cell_id, category, wsI);
    k_logits<<<CC*MT, 32, 0, stream>>>(vec, Wlin, blin, true_idx, wsI, out);
    k_ecrel <<<NN/8, 256, 0, stream>>>(vec, Wec, bec, Wrel, brel, ec_idx, rel_idx, out);
    k_lstm  <<<KK*MT32, 256, 0, stream>>>(inp, h_prev, c_prev, W_ih, b_ih, W_hh, b_hh, wsI, out);
}
